// _Separable1_75909251989661
// MI455X (gfx1250) — matrix operand order repaired and checked by reading, not compiled, not run on hardware
//
#include <hip/hip_runtime.h>
#include <hip/hip_bf16.h>

// Problem dims (compile-time)
#define BDIM 2
#define CCH 32
#define WDIM 96
#define HDIM 96
#define ZDIM 64
#define WH (WDIM*HDIM)            // 9216
#define CHSTR (WH*ZDIM)           // 589824 elems per (b,c) channel slab
#define NROWS (BDIM*WH)           // 18432 attention rows
#define NOUT (BDIM*CCH*CHSTR)     // 37748736 output elems
#define BN_N ((float)(BDIM*(size_t)WH*ZDIM)) // 1179648 per-channel count

#if defined(__has_builtin)
#  if __has_builtin(__builtin_amdgcn_global_load_async_to_lds_b64)
#    define HAVE_ASYNC_LDS 1
#  else
#    define HAVE_ASYNC_LDS 0
#  endif
#else
#  define HAVE_ASYNC_LDS 0
#endif

typedef __attribute__((ext_vector_type(16))) __bf16   v16bf;
typedef __attribute__((ext_vector_type(8)))  float    v8f;
typedef __attribute__((ext_vector_type(8)))  unsigned v8u;
typedef __attribute__((ext_vector_type(2)))  int      v2i;
// address-space qualified pointers for the async builtin
typedef __attribute__((address_space(1))) v2i* gv2ip;  // global
typedef __attribute__((address_space(3))) v2i* lv2ip;  // LDS

// ---- bf16 helpers -----------------------------------------------------------
// RNE (one-time weight conversion)
static __device__ inline unsigned short f2bf_rne(float f){
  unsigned u = __builtin_bit_cast(unsigned, f);
  u += 0x7FFFu + ((u >> 16) & 1u);
  return (unsigned short)(u >> 16);
}
static __device__ inline unsigned pk2_rne(float a, float b){
  return (unsigned)f2bf_rne(a) | ((unsigned)f2bf_rne(b) << 16);
}
// RTZ truncation (hot loop): single v_perm_b32 per pair
static __device__ inline unsigned pk2(float a, float b){
  return __builtin_amdgcn_perm(__builtin_bit_cast(unsigned, b),
                               __builtin_bit_cast(unsigned, a), 0x07060302u);
}
// RTZ bf16 store: compiles to ds_store_b16_d16_hi (no ALU)
static __device__ inline void st_bf(unsigned short* p, float v){
  *p = (unsigned short)(__builtin_bit_cast(unsigned, v) >> 16);
}
static __device__ inline v8f vzero(){
  v8f z;
  #pragma unroll
  for (int i = 0; i < 8; ++i) z[i] = 0.f;
  return z;
}
static __device__ inline v8f wmma_bf16(v8u a, v8u b, v8f c){
  return __builtin_amdgcn_wmma_f32_16x16x32_bf16(
      false, __builtin_bit_cast(v16bf, a),
      false, __builtin_bit_cast(v16bf, b),
      (short)0, c, false, false);
}
static __device__ inline void lds_fence(){
  asm volatile("s_wait_dscnt 0" ::: "memory");
}
static __device__ inline void async_fence(){
#if HAVE_ASYNC_LDS
#  if __has_builtin(__builtin_amdgcn_s_wait_asynccnt)
  __builtin_amdgcn_s_wait_asynccnt(0);
#  else
  asm volatile("s_wait_asynccnt 0" ::: "memory");
#  endif
#endif
}

// Per-wave LDS region layout (26624 B per wave, 2 waves/block):
//   [    0, 8192) XF  : x row, f32  [c][z]  (32x64)
//   [ 8192,12288) Tb  : theta(x),   bf16 [o][z]
//   [12288,16384) Pb  : phi(x),     bf16 [o][z]
//   [16384,20480) GXb : g(x),       bf16 [c][z]
//   [20480,22528) Ab  : attn tile,  bf16 [m][y] (16x64)
//   [22528,26624) Rb  : residual,   bf16 [c][z]
#define WAVE_LDS 26624

__global__ __launch_bounds__(64) void fused_nl_kernel(
    const float* __restrict__ x,       const float* __restrict__ spacings,
    const float* __restrict__ theta_w, const float* __restrict__ theta_b,
    const float* __restrict__ phi_w,   const float* __restrict__ phi_b,
    const float* __restrict__ g_w,     const float* __restrict__ g_b,
    const float* __restrict__ Ww,      const float* __restrict__ Wbias,
    float* __restrict__ out, float* __restrict__ stats)
{
  extern __shared__ char smem[];
  const int tid  = threadIdx.x;
  const int wave = tid >> 5;
  const int lane = tid & 31;
  const int lh   = lane >> 4;   // lane half (K-group / M+8 select)
  const int ln   = lane & 15;

  char* base = smem + wave * WAVE_LDS;
  float*          XF  = (float*)(base);
  unsigned short* Tb  = (unsigned short*)(base + 8192);
  unsigned short* Pb  = (unsigned short*)(base + 12288);
  unsigned short* GXb = (unsigned short*)(base + 16384);
  unsigned short* Ab  = (unsigned short*)(base + 20480);
  unsigned short* Rb  = (unsigned short*)(base + 22528);

  // ---- weight A-fragments (bf16 16x32 layout), persist across rows ----
  // A layout: lane row M = lane%16; V0..7 -> K = 2v + 16*lh
  v8u thA[2], phA[2], wA[2];
  #pragma unroll
  for (int mt = 0; mt < 2; ++mt){
    const int m = mt*16 + ln;
    #pragma unroll
    for (int v = 0; v < 8; ++v){
      const int k = 2*v + 16*lh;
      thA[mt][v] = pk2_rne(theta_w[m*32+k], theta_w[m*32+k+1]);
      phA[mt][v] = pk2_rne(phi_w  [m*32+k], phi_w  [m*32+k+1]);
      wA [mt][v] = pk2_rne(Ww     [m*32+k], Ww     [m*32+k+1]);
    }
  }
  // Biases per C/D-fragment element: channel = mt*16 + r + 8*lh
  float thb[2][8], phb[2][8], wbb[2][8];
  #pragma unroll
  for (int mt = 0; mt < 2; ++mt)
    #pragma unroll
    for (int r = 0; r < 8; ++r){
      const int o = mt*16 + r + 8*lh;
      thb[mt][r] = theta_b[o]; phb[mt][r] = phi_b[o]; wbb[mt][r] = Wbias[o];
    }

  // BN partial sums (per-lane; channel = mt*16 + r + 8*lh, N-subset = this lane's z's)
  float ps[2][8], pq[2][8];
  #pragma unroll
  for (int mt = 0; mt < 2; ++mt)
    #pragma unroll
    for (int r = 0; r < 8; ++r){ ps[mt][r] = 0.f; pq[mt][r] = 0.f; }

  for (int row = blockIdx.x*2 + wave; row < NROWS; row += gridDim.x*2)
  {
    const int b  = row / WH;
    const int wh = row % WH;
    const int useNL = spacings[b*3 + 2] <= 1.5f;    // per-sample z-spacing branch
    const float* xrow = x + (size_t)b*CCH*CHSTR + (size_t)wh*ZDIM;

    // 1) stage x row to LDS (f32), 8B/lane contiguous per channel
#if HAVE_ASYNC_LDS
    #pragma unroll 4
    for (int c = 0; c < 32; ++c){
      __builtin_amdgcn_global_load_async_to_lds_b64(
          (gv2ip)(xrow + (size_t)c*CHSTR + 2*lane),
          (lv2ip)(XF + c*64 + 2*lane),
          0, 0);
    }
    async_fence();
#else
    #pragma unroll 4
    for (int c = 0; c < 32; ++c){
      const float2 v = *(const float2*)(xrow + (size_t)c*CHSTR + 2*lane);
      *(float2*)(XF + c*64 + 2*lane) = v;
    }
    lds_fence();
#endif

    // 2) depthwise 3-tap conv along z (pad 1) -> GXb (bf16)
    #pragma unroll 4
    for (int c = 0; c < 32; ++c){
      const float w0 = g_w[c*3+0], w1 = g_w[c*3+1], w2 = g_w[c*3+2], bb = g_b[c];
      const int z0 = 2*lane;
      const float xm = z0 ? XF[c*64 + z0-1] : 0.f;
      const float x0 = XF[c*64 + z0];
      const float x1 = XF[c*64 + z0+1];
      const float xp = (z0+2 < 64) ? XF[c*64 + z0+2] : 0.f;
      ((unsigned*)GXb)[c*32 + lane] = pk2(w0*xm + w1*x0 + w2*x1 + bb,
                                          w0*x0 + w1*x1 + w2*xp + bb);
    }

    // 3) t = theta·x + b, p = phi·x + b   (M=o 32, N=z 64, K=c 32)
    #pragma unroll
    for (int nt = 0; nt < 4; ++nt){
      v8u bx;  // B layout: N = nt*16+ln, V_v holds K = 2v+16*lh, +1
      #pragma unroll
      for (int v = 0; v < 8; ++v){
        const int k = 2*v + 16*lh;
        bx[v] = pk2(XF[k*64 + nt*16 + ln], XF[(k+1)*64 + nt*16 + ln]);
      }
      #pragma unroll
      for (int mt = 0; mt < 2; ++mt){
        v8f t = wmma_bf16(thA[mt], bx, vzero());
        v8f p = wmma_bf16(phA[mt], bx, vzero());
        #pragma unroll
        for (int r = 0; r < 8; ++r){
          const int o = mt*16 + r + 8*lh;
          st_bf(&Tb[o*64 + nt*16 + ln], t[r] + thb[mt][r]);
          st_bf(&Pb[o*64 + nt*16 + ln], p[r] + phb[mt][r]);
        }
      }
    }
    lds_fence();

    // 4) attention per 16-wide z-tile: f = t^T p ; softmax(y) ; y = a · gx^T
    #pragma unroll
    for (int zt = 0; zt < 4; ++zt){
      v8u at;  // A = t^T : M = z (tile), K = o
      #pragma unroll
      for (int v = 0; v < 8; ++v){
        const int k = 2*v + 16*lh;
        const int m = zt*16 + ln;
        at[v] = (unsigned)Tb[k*64 + m] | ((unsigned)Tb[(k+1)*64 + m] << 16);
      }
      v8f fa[4];
      #pragma unroll
      for (int yt = 0; yt < 4; ++yt){
        v8u bp;  // B = p : K = o, N = y
        #pragma unroll
        for (int v = 0; v < 8; ++v){
          const int k = 2*v + 16*lh;
          bp[v] = (unsigned)Pb[k*64 + yt*16 + ln] |
                 ((unsigned)Pb[(k+1)*64 + yt*16 + ln] << 16);
        }
        fa[yt] = wmma_bf16(at, bp, vzero());
      }
      // softmax over y (row M = r+8*lh lives in one 16-lane half across 4 frags)
      #pragma unroll
      for (int r = 0; r < 8; ++r){
        float mx = fmaxf(fmaxf(fa[0][r], fa[1][r]), fmaxf(fa[2][r], fa[3][r]));
        #pragma unroll
        for (int d = 1; d < 16; d <<= 1) mx = fmaxf(mx, __shfl_xor(mx, d, 32));
        float s = 0.f;
        #pragma unroll
        for (int yt = 0; yt < 4; ++yt){
          const float e = __expf(fa[yt][r] - mx); fa[yt][r] = e; s += e;
        }
        #pragma unroll
        for (int d = 1; d < 16; d <<= 1) s += __shfl_xor(s, d, 32);
        const float inv = 1.0f / s;
        #pragma unroll
        for (int yt = 0; yt < 4; ++yt)
          st_bf(&Ab[(r + 8*lh)*64 + yt*16 + ln], fa[yt][r] * inv);
      }
      lds_fence();
      // y_tile(16z x 32c) = a(16x64) x gx^T(64x32), K = y in two 32-steps
      v8f ya[2] = { vzero(), vzero() };
      #pragma unroll
      for (int ks = 0; ks < 2; ++ks){
        v8u aa;  // A from Ab row-major, K-pairs contiguous -> dword loads
        #pragma unroll
        for (int v = 0; v < 8; ++v){
          const int k = 2*v + 16*lh + 32*ks;
          aa[v] = *(const unsigned*)(Ab + ln*64 + k);
        }
        #pragma unroll
        for (int nt = 0; nt < 2; ++nt){
          v8u bg;  // B[k=y][n=c] = GXb[c*64+y], y-pairs contiguous
          #pragma unroll
          for (int v = 0; v < 8; ++v){
            const int k = 2*v + 16*lh + 32*ks;
            bg[v] = *(const unsigned*)(GXb + (nt*16 + ln)*64 + k);
          }
          ya[nt] = wmma_bf16(aa, bg, ya[nt]);
        }
      }
      // residual + per-sample branch; scatter transposed into Rb[c][z]
      #pragma unroll
      for (int nt = 0; nt < 2; ++nt)
        #pragma unroll
        for (int r = 0; r < 8; ++r){
          const int c = nt*16 + ln;
          const int z = zt*16 + r + 8*lh;
          const float xv = XF[c*64 + z];
          st_bf(&Rb[c*64 + z], useNL ? (ya[nt][r] + xv) : xv);
        }
    }
    lds_fence();

    // 5) out = Ww·res + Wb ; store + accumulate BN partials
    #pragma unroll
    for (int nt = 0; nt < 4; ++nt){
      v8u br;
      #pragma unroll
      for (int v = 0; v < 8; ++v){
        const int k = 2*v + 16*lh;
        br[v] = (unsigned)Rb[k*64 + nt*16 + ln] |
               ((unsigned)Rb[(k+1)*64 + nt*16 + ln] << 16);
      }
      #pragma unroll
      for (int mt = 0; mt < 2; ++mt){
        v8f o = wmma_bf16(wA[mt], br, vzero());
        #pragma unroll
        for (int r = 0; r < 8; ++r){
          const int co = mt*16 + r + 8*lh;
          const int z  = nt*16 + ln;
          const float val = o[r] + wbb[mt][r];
          out[(size_t)(b*CCH + co)*CHSTR + (size_t)wh*ZDIM + z] = val;
          ps[mt][r] += val;
          pq[mt][r] += val*val;
        }
      }
    }
  }

  // 6) half-wave reduce BN partials, one atomic per channel per wave
  #pragma unroll
  for (int mt = 0; mt < 2; ++mt)
    #pragma unroll
    for (int r = 0; r < 8; ++r){
      float s = ps[mt][r], q = pq[mt][r];
      #pragma unroll
      for (int d = 1; d < 16; d <<= 1){
        s += __shfl_xor(s, d, 32);
        q += __shfl_xor(q, d, 32);
      }
      if (ln == 0){
        const int co = mt*16 + r + 8*lh;
        atomicAdd(&stats[co], s);
        atomicAdd(&stats[32 + co], q);
      }
    }
}

__global__ void zero_stats_kernel(float* stats){
  if (threadIdx.x < 64) stats[threadIdx.x] = 0.f;
}

__global__ void bn_finalize_kernel(float* stats, const float* __restrict__ gamma,
                                   const float* __restrict__ beta){
  const int c = threadIdx.x;
  if (c < 32){
    const float n    = BN_N;
    const float mean = stats[c] / n;
    const float var  = stats[32 + c] / n - mean*mean;
    const float sc   = gamma[c] * rsqrtf(var + 1e-5f);
    stats[64 + c] = sc;
    stats[96 + c] = beta[c] - mean*sc;
  }
}

__global__ void bn_apply_kernel(float* __restrict__ out, const float* __restrict__ stats){
  __shared__ float sc[32], sh[32];
  if (threadIdx.x < 32){
    sc[threadIdx.x] = stats[64 + threadIdx.x];
    sh[threadIdx.x] = stats[96 + threadIdx.x];
  }
  __syncthreads();
  for (size_t i = (size_t)blockIdx.x*blockDim.x + threadIdx.x;
       i < (size_t)NOUT; i += (size_t)gridDim.x*blockDim.x){
    const int c = (int)((i / CHSTR) & 31);
    out[i] = out[i]*sc[c] + sh[c];
  }
}

extern "C" void kernel_launch(void* const* d_in, const int* in_sizes, int n_in,
                              void* d_out, int out_size, void* d_ws, size_t ws_size,
                              hipStream_t stream){
  const float* x        = (const float*)d_in[0];
  const float* spacings = (const float*)d_in[1];
  const float* theta_w  = (const float*)d_in[2];
  const float* theta_b  = (const float*)d_in[3];
  const float* phi_w    = (const float*)d_in[4];
  const float* phi_b    = (const float*)d_in[5];
  const float* g_w      = (const float*)d_in[6];
  const float* g_b      = (const float*)d_in[7];
  const float* W_w      = (const float*)d_in[8];
  const float* W_b      = (const float*)d_in[9];
  const float* bn_gamma = (const float*)d_in[10];
  const float* bn_beta  = (const float*)d_in[11];
  float* out   = (float*)d_out;
  float* stats = (float*)d_ws;   // [0:32) sum, [32:64) sumsq, [64:96) scale, [96:128) shift

  zero_stats_kernel<<<1, 64, 0, stream>>>(stats);
  // 2304 blocks x 2 waves = 4608 waves, 4 rows each (18432 rows total)
  fused_nl_kernel<<<2304, 64, 2*WAVE_LDS, stream>>>(
      x, spacings, theta_w, theta_b, phi_w, phi_b,
      g_w, g_b, W_w, W_b, out, stats);
  bn_finalize_kernel<<<1, 32, 0, stream>>>(stats, bn_gamma, bn_beta);
  bn_apply_kernel<<<4096, 256, 0, stream>>>(out, stats);
}
